// Attn_8186207666920
// MI455X (gfx1250) — matrix operand order repaired and checked by reading, not compiled, not run on hardware
//
#include <hip/hip_runtime.h>

// ---------------------------------------------------------------------------
// Types / constants
// ---------------------------------------------------------------------------
typedef __attribute__((ext_vector_type(16))) __bf16          v16bf;
typedef __attribute__((ext_vector_type(8)))  __bf16          v8bf;
typedef __attribute__((ext_vector_type(8)))  float           v8f;
typedef __attribute__((ext_vector_type(4)))  unsigned int    v4u32;
typedef __attribute__((ext_vector_type(4)))  unsigned int    u32x4;
typedef __attribute__((ext_vector_type(4)))  int             i32x4;
typedef __attribute__((ext_vector_type(8)))  int             i32x8;

#define WMMA_BF16(a, b, c)                                                    \
  __builtin_amdgcn_wmma_f32_16x16x32_bf16(false, (a), false, (b), (short)0,   \
                                          (c), false, false)

static constexpr int   kB      = 2;
static constexpr int   kT      = 2048;
static constexpr int   kDM     = 2048;
static constexpr int   kNH     = 16;
static constexpr int   kDH     = 128;
static constexpr int   kM      = kB * kT;    // 4096 rows (b,t flattened)
static constexpr int   kN1     = 3 * kDM;    // 6144 qkv cols
static constexpr float kLog2e  = 1.4426950408889634f;
static constexpr float kNegBig = -3.0e38f;

// hardware bf16 convert (v_cvt_pk_bf16_f32 on gfx1250)
__device__ __forceinline__ unsigned short f2bf(float f) {
  return __builtin_bit_cast(unsigned short, (__bf16)f);
}
__device__ __forceinline__ float bf2f(unsigned short s) {
  return __uint_as_float(((unsigned int)s) << 16);
}
// 8 x f32 -> 8 x bf16 packed; convertvector lets the backend use native
// v_cvt_pk_bf16_f32 pairing without perm fixups.
__device__ __forceinline__ v4u32 pack8(float4 a, float4 b) {
  v8f f;
  f[0] = a.x; f[1] = a.y; f[2] = a.z; f[3] = a.w;
  f[4] = b.x; f[5] = b.y; f[6] = b.z; f[7] = b.w;
  return __builtin_bit_cast(v4u32, __builtin_convertvector(f, v8bf));
}
__device__ __forceinline__ v4u32 pack8v(v8f f) {
  return __builtin_bit_cast(v4u32, __builtin_convertvector(f, v8bf));
}

// Two 16-byte loads -> one 16-bit-element fragment (b128 codegen).
struct u32x8s { v4u32 lo, hi; };
__device__ __forceinline__ v16bf frag_from(const unsigned short* lo,
                                           const unsigned short* hi) {
  u32x8s t;
  t.lo = *(const v4u32*)lo;
  t.hi = *(const v4u32*)hi;
  return __builtin_bit_cast(v16bf, t);
}

// ---------------------------------------------------------------------------
// CDNA5 async global->LDS copy (ASYNCcnt), 16B per lane.
// LDS offset = low 32 bits of the generic address (flat->LDS mapping).
// ---------------------------------------------------------------------------
__device__ __forceinline__ void async_copy16(void* lds_dst, const void* gsrc) {
  unsigned off          = (unsigned)(uintptr_t)lds_dst;
  unsigned long long ga = (unsigned long long)(uintptr_t)gsrc;
  asm volatile("global_load_async_to_lds_b128 %0, %1, off"
               :: "v"(off), "v"(ga) : "memory");
}
__device__ __forceinline__ void wait_async0() {
  asm volatile("s_wait_asynccnt 0x0" ::: "memory");
}

// ---------------------------------------------------------------------------
// CDNA5 Tensor Data Mover: 2D tile load (D# groups per ISA 08_async_tensor §8)
// ---------------------------------------------------------------------------
#if __has_builtin(__builtin_amdgcn_tensor_load_to_lds)
#define HAVE_TDM 1
__device__ __forceinline__ void set_bits(unsigned* a, int off, int width,
                                         unsigned long long v) {
  int i = off >> 5, sh = off & 31;
  a[i] |= (unsigned)(v << sh);
  if (sh + width > 32) a[i + 1] |= (unsigned)(v >> (32 - sh));
}
// Load tile (tile_d0 x tile_d1) of 2-byte elements from a 2D tensor with row
// stride row_stride (elements) into LDS at lds_dst (linear, row-major tile).
__device__ __forceinline__ void tdm_load_2d(void* lds_dst, const void* gsrc,
                                            unsigned tile_d0, unsigned tile_d1,
                                            unsigned long long row_stride,
                                            unsigned tensor_d0,
                                            unsigned tensor_d1) {
  unsigned long long ga = (unsigned long long)(uintptr_t)gsrc;
  u32x4 g0;
  g0[0] = 1u;                                  // count=1, user descriptor
  g0[1] = (unsigned)(uintptr_t)lds_dst;        // lds_addr (bytes)
  g0[2] = (unsigned)(ga & 0xFFFFFFFFu);        // global_addr[31:0]
  g0[3] = (unsigned)((ga >> 32) & 0x1FFFFFFu) | (2u << 30);  // [56:32]|type=2
  unsigned w[8] = {0, 0, 0, 0, 0, 0, 0, 0};
  set_bits(w, 16, 2, 1);                       // data_size: 1 -> 2 bytes
  set_bits(w, 48, 32, tensor_d0);              // tensor_dim0
  set_bits(w, 80, 32, tensor_d1);              // tensor_dim1
  set_bits(w, 112, 16, tile_d0);               // tile_dim0
  set_bits(w, 128, 16, tile_d1);               // tile_dim1
  set_bits(w, 160, 48, row_stride);            // tensor_dim0_stride
  i32x8 g1;
#pragma unroll
  for (int i = 0; i < 8; ++i) g1[i] = (int)w[i];
  i32x4 z4 = {0, 0, 0, 0};
#if __clang_major__ >= 23
  i32x8 z8 = {0, 0, 0, 0, 0, 0, 0, 0};
  __builtin_amdgcn_tensor_load_to_lds(g0, g1, z4, z4, z8, 0);
#else
  __builtin_amdgcn_tensor_load_to_lds(g0, g1, z4, z4, 0);
#endif
}
#endif  // HAVE_TDM

// ---------------------------------------------------------------------------
// WMMA fragment loaders (wave32 layouts, 16-bit operands, K=32)
// ---------------------------------------------------------------------------
__device__ __forceinline__ v16bf load_a_frag(const unsigned short* src, int srs,
                                             int lane) {
  const unsigned short* p = src + (lane & 15) * srs + ((lane >> 4) << 3);
  return frag_from(p, p + 16);
}
// B[k][n] sourced from n-major storage src[n][k] (row stride srs)
__device__ __forceinline__ v16bf load_b_frag_nmajor(const unsigned short* src,
                                                    int srs, int lane) {
  const unsigned short* p = src + (lane & 15) * srs + ((lane >> 4) << 3);
  return frag_from(p, p + 16);
}

// ---------------------------------------------------------------------------
// Kernel 1: qkv = x @ w_qkv^T  (f32 in, bf16 out scattered per-head;
//           v written TRANSPOSED [b][h][d][t] for the attention P*V path)
// ---------------------------------------------------------------------------
__global__ __launch_bounds__(256) void qkv_gemm_kernel(
    const float* __restrict__ x, const float* __restrict__ w,
    unsigned short* __restrict__ qh, unsigned short* __restrict__ kh,
    unsigned short* __restrict__ vt) {
  __shared__ __align__(16) unsigned short As[64 * 32];
  __shared__ __align__(16) unsigned short Bs[128 * 32];

  const int tid  = threadIdx.x;
  const int lane = tid & 31;
  const int wave = tid >> 5;
  const int wm   = wave >> 2;
  const int wn   = wave & 3;
  const int nb   = blockIdx.x * 128;
  const int mb   = blockIdx.y * 64;

  const int ar = tid >> 2, ac = (tid & 3) * 8;
  const int br = tid >> 1, bc = (tid & 1) * 16;
  const float* xrow = x + (size_t)(mb + ar) * kDM + ac;
  const float* wrow = w + (size_t)(nb + br) * kDM + bc;

  v8f acc[2][2] = {};

  for (int kb = 0; kb < kDM; kb += 32) {
    const float4* xa = (const float4*)(xrow + kb);
    const float4* wb = (const float4*)(wrow + kb);
    float4 a0 = xa[0], a1 = xa[1];
    float4 b0 = wb[0], b1 = wb[1], b2 = wb[2], b3 = wb[3];
    *(v4u32*)&As[ar * 32 + ac]     = pack8(a0, a1);
    *(v4u32*)&Bs[br * 32 + bc]     = pack8(b0, b1);
    *(v4u32*)&Bs[br * 32 + bc + 8] = pack8(b2, b3);
    if (kb + 32 < kDM) {
      __builtin_prefetch(xrow + kb + 32, 0, 3);
      __builtin_prefetch(wrow + kb + 32, 0, 3);
    }
    __syncthreads();

    v16bf fa0 = load_a_frag(As + (wm * 32 + 0) * 32, 32, lane);
    v16bf fa1 = load_a_frag(As + (wm * 32 + 16) * 32, 32, lane);
    v16bf fb0 = load_b_frag_nmajor(Bs + (wn * 32 + 0) * 32, 32, lane);
    v16bf fb1 = load_b_frag_nmajor(Bs + (wn * 32 + 16) * 32, 32, lane);
    acc[0][0] = WMMA_BF16(fa0, fb0, acc[0][0]);
    acc[0][1] = WMMA_BF16(fa0, fb1, acc[0][1]);
    acc[1][0] = WMMA_BF16(fa1, fb0, acc[1][0]);
    acc[1][1] = WMMA_BF16(fa1, fb1, acc[1][1]);
    __syncthreads();
  }

  const int hi = lane >> 4;
#pragma unroll
  for (int sm = 0; sm < 2; ++sm)
#pragma unroll
    for (int sn = 0; sn < 2; ++sn) {
      int n     = nb + wn * 32 + sn * 16 + (lane & 15);
      int which = n / kDM;
      int c     = n % kDM;
      int h     = c >> 7, d = c & 127;
      int m0    = mb + wm * 32 + sm * 16 + 8 * hi;
      int b     = m0 >> 11, t0 = m0 & (kT - 1);
      if (which == 2) {
        // v transposed: 8 consecutive t -> one b128 store
        *(v4u32*)&vt[((size_t)(b * kNH + h) * kDH + d) * kT + t0] =
            pack8v(acc[sm][sn]);
      } else {
        unsigned short* dst = which ? kh : qh;
#pragma unroll
        for (int r = 0; r < 8; ++r)
          dst[((size_t)(b * kNH + h) * kT + t0 + r) * kDH + d] =
              f2bf(acc[sm][sn][r]);
      }
    }
}

// ---------------------------------------------------------------------------
// Kernel 2: RoPE in place on q (1/sqrt(DH) folded in) and k.
// ---------------------------------------------------------------------------
__global__ __launch_bounds__(256) void rope_kernel(
    unsigned short* __restrict__ qh, unsigned short* __restrict__ kh) {
  int idx = blockIdx.x * blockDim.x + threadIdx.x;   // < B*NH*T*64 = 2^22
  int d   = idx & 63;
  int t   = (idx >> 6) & (kT - 1);
  int bh  = idx >> 17;
  unsigned short* buf = (blockIdx.y == 0) ? qh : kh;
  float scale = (blockIdx.y == 0) ? 0.08838834764831845f : 1.0f;  // 1/sqrt(128)

  size_t base = (size_t)bh * kT * kDH + (size_t)t * kDH;
  float invf = exp2f(-(float)(2 * d) * (13.287712379549449f / 128.0f));
  float fr = (float)t * invf;
  float sn, cs;
  __sincosf(fr, &sn, &cs);
  float a1 = bf2f(buf[base + d]);
  float a2 = bf2f(buf[base + d + 64]);
  buf[base + d]      = f2bf((a1 * cs - a2 * sn) * scale);
  buf[base + d + 64] = f2bf((a2 * cs + a1 * sn) * scale);
}

// ---------------------------------------------------------------------------
// Kernel 3: causal flash attention, one block per (b, h, 64 q rows).
//   K/V tiles staged by the Tensor Data Mover (TENSORcnt) when available,
//   otherwise by async global->LDS loads (ASYNCcnt).
// ---------------------------------------------------------------------------
__global__ __launch_bounds__(128) void flash_attn_kernel(
    const unsigned short* __restrict__ qh, const unsigned short* __restrict__ kh,
    const unsigned short* __restrict__ vt, unsigned short* __restrict__ ao) {
  __shared__ __align__(16) unsigned short Ks[32 * 128];  // [key][d]
  __shared__ __align__(16) unsigned short Vs[128 * 32];  // [d][key]
  __shared__ __align__(16) unsigned short Ps[4 * 16 * 32];

  const int tid  = threadIdx.x;
  const int lane = tid & 31;
  const int wq   = tid >> 5;
  const int hi   = lane >> 4;
  const int qb   = blockIdx.x * 64;
  const int h    = blockIdx.y;
  const int b    = blockIdx.z;

  const size_t headoff = (size_t)(b * kNH + h) * kT * kDH;
  const unsigned short* qptr = qh + headoff;
  const char* kbytes = (const char*)(kh + headoff);
  const char* vbytes = (const char*)(vt + headoff);

  // Q held in registers: 16x128 as 4 A-fragments (K=32 each)
  v16bf qf[4];
  {
    const unsigned short* p =
        qptr + (size_t)(qb + wq * 16 + (lane & 15)) * kDH + ((lane >> 4) << 3);
#pragma unroll
    for (int dsl = 0; dsl < 4; ++dsl)
      qf[dsl] = frag_from(p + dsl * 32, p + dsl * 32 + 16);
  }

  float mrow[8], lrow[8], alpha[8];
  v8f o[8] = {};
#pragma unroll
  for (int r = 0; r < 8; ++r) { mrow[r] = kNegBig; lrow[r] = 0.0f; }

  const int nkb = (qb + 64) / 32;
  for (int kb = 0; kb < nkb; ++kb) {
    const int j0 = kb * 32;

#if HAVE_TDM
    if (wq == 0) {   // one wave drives the TDM; EXEC is irrelevant for TENSOR
      // K tile: 32 rows (keys) x 128 cols (d), row stride 128 elems
      tdm_load_2d(Ks, kbytes + (size_t)j0 * (kDH * 2), kDH, 32,
                  (unsigned long long)kDH, kDH, kT);
      // V tile (transposed tensor [d][t]): 128 rows x 32 cols, stride kT
      tdm_load_2d(Vs, vbytes + (size_t)j0 * 2, 32, kDH,
                  (unsigned long long)kT, kT, kDH);
      __builtin_amdgcn_s_wait_tensorcnt(0);
    }
#else
    {
      const char* ksrc = kbytes + (size_t)j0 * (kDH * 2);
      const char* vsrc = vbytes + (size_t)tid * (kT * 2) + (size_t)j0 * 2;
      char* kdst = (char*)Ks;
      char* vdst = (char*)Vs + tid * 64;
#pragma unroll
      for (int i = 0; i < 4; ++i) {
        async_copy16(kdst + tid * 16 + i * 2048, ksrc + tid * 16 + i * 2048);
        async_copy16(vdst + i * 16, vsrc + i * 16);
      }
      wait_async0();
    }
#endif
    __syncthreads();

    // S = Q * K^T, two 16-key tiles, DH = 4 x 32
    v8f s0 = {}, s1 = {};
#pragma unroll
    for (int dsl = 0; dsl < 4; ++dsl) {
      v16bf bk0 = load_b_frag_nmajor(Ks + 0 * 128 + dsl * 32, 128, lane);
      v16bf bk1 = load_b_frag_nmajor(Ks + 16 * 128 + dsl * 32, 128, lane);
      s0 = WMMA_BF16(qf[dsl], bk0, s0);
      s1 = WMMA_BF16(qf[dsl], bk1, s1);
    }

    // online softmax (per-row stats across the 16-lane N group)
#pragma unroll
    for (int r = 0; r < 8; ++r) {
      int q   = qb + wq * 16 + r + 8 * hi;
      int k0g = j0 + (lane & 15);
      float a0 = (k0g <= q) ? s0[r] : kNegBig;
      float a1 = (k0g + 16 <= q) ? s1[r] : kNegBig;
      float mx = fmaxf(a0, a1);
#pragma unroll
      for (int off = 1; off < 16; off <<= 1)
        mx = fmaxf(mx, __shfl_xor(mx, off, 32));
      float mnew = fmaxf(mrow[r], mx);
      alpha[r]  = exp2f((mrow[r] - mnew) * kLog2e);
      float p0  = exp2f((a0 - mnew) * kLog2e);
      float p1  = exp2f((a1 - mnew) * kLog2e);
      float rs  = p0 + p1;
#pragma unroll
      for (int off = 1; off < 16; off <<= 1)
        rs += __shfl_xor(rs, off, 32);
      lrow[r] = lrow[r] * alpha[r] + rs;
      mrow[r] = mnew;
      unsigned short* pw = Ps + wq * 512 + (r + 8 * hi) * 32;
      pw[(lane & 15)]      = f2bf(p0);
      pw[16 + (lane & 15)] = f2bf(p1);
    }
#pragma unroll
    for (int dsl = 0; dsl < 8; ++dsl)
#pragma unroll
      for (int r = 0; r < 8; ++r) o[dsl][r] *= alpha[r];

    // O += P * V   (V already [d][key] in LDS -> contiguous B-frag reads)
    v16bf pf = load_a_frag(Ps + wq * 512, 32, lane);
#pragma unroll
    for (int dsl = 0; dsl < 8; ++dsl) {
      v16bf bv = load_b_frag_nmajor(Vs + dsl * 16 * 32, 32, lane);
      o[dsl] = WMMA_BF16(pf, bv, o[dsl]);
    }
    __syncthreads();
  }

  // normalize and write ao[b][t][h*DH + d] (bf16)
#pragma unroll
  for (int dsl = 0; dsl < 8; ++dsl) {
    int d = dsl * 16 + (lane & 15);
#pragma unroll
    for (int r = 0; r < 8; ++r) {
      int t = qb + wq * 16 + r + 8 * hi;
      ao[((size_t)(b * kT + t)) * kDM + h * kDH + d] =
          f2bf(o[dsl][r] / lrow[r]);
    }
  }
}

// ---------------------------------------------------------------------------
// Kernel 4: out = ao @ w_op^T  (bf16 x bf16 -> f32); A tile async-staged.
// ---------------------------------------------------------------------------
__global__ __launch_bounds__(256) void out_proj_kernel(
    const unsigned short* __restrict__ ao, const float* __restrict__ w,
    float* __restrict__ out) {
  __shared__ __align__(16) unsigned short As[64 * 32];
  __shared__ __align__(16) unsigned short Bs[128 * 32];

  const int tid  = threadIdx.x;
  const int lane = tid & 31;
  const int wave = tid >> 5;
  const int wm   = wave >> 2;
  const int wn   = wave & 3;
  const int nb   = blockIdx.x * 128;
  const int mb   = blockIdx.y * 64;

  const char* arow =
      (const char*)(ao + (size_t)(mb + (tid >> 2)) * kDM + (tid & 3) * 8);
  const int br = tid >> 1, bc = (tid & 1) * 16;
  const float* wrow = w + (size_t)(nb + br) * kDM + bc;

  v8f acc[2][2] = {};

  for (int kb = 0; kb < kDM; kb += 32) {
    async_copy16((char*)As + tid * 16, arow + (size_t)kb * 2);
    const float4* wb = (const float4*)(wrow + kb);
    float4 b0 = wb[0], b1 = wb[1], b2 = wb[2], b3 = wb[3];
    *(v4u32*)&Bs[br * 32 + bc]     = pack8(b0, b1);
    *(v4u32*)&Bs[br * 32 + bc + 8] = pack8(b2, b3);
    if (kb + 32 < kDM) {
      __builtin_prefetch(arow + (size_t)(kb + 32) * 2, 0, 3);
      __builtin_prefetch(wrow + kb + 32, 0, 3);
    }
    wait_async0();
    __syncthreads();

    v16bf fa0 = load_a_frag(As + (wm * 32 + 0) * 32, 32, lane);
    v16bf fa1 = load_a_frag(As + (wm * 32 + 16) * 32, 32, lane);
    v16bf fb0 = load_b_frag_nmajor(Bs + (wn * 32 + 0) * 32, 32, lane);
    v16bf fb1 = load_b_frag_nmajor(Bs + (wn * 32 + 16) * 32, 32, lane);
    acc[0][0] = WMMA_BF16(fa0, fb0, acc[0][0]);
    acc[0][1] = WMMA_BF16(fa0, fb1, acc[0][1]);
    acc[1][0] = WMMA_BF16(fa1, fb0, acc[1][0]);
    acc[1][1] = WMMA_BF16(fa1, fb1, acc[1][1]);
    __syncthreads();
  }

  const int hi = lane >> 4;
#pragma unroll
  for (int sm = 0; sm < 2; ++sm)
#pragma unroll
    for (int sn = 0; sn < 2; ++sn) {
      int n = nb + wn * 32 + sn * 16 + (lane & 15);
#pragma unroll
      for (int r = 0; r < 8; ++r) {
        int m = mb + wm * 32 + sm * 16 + r + 8 * hi;
        out[(size_t)m * kDM + n] = acc[sm][sn][r];
      }
    }
}

// ---------------------------------------------------------------------------
// Launch
// ---------------------------------------------------------------------------
extern "C" void kernel_launch(void* const* d_in, const int* in_sizes, int n_in,
                              void* d_out, int out_size, void* d_ws,
                              size_t ws_size, hipStream_t stream) {
  const float* x     = (const float*)d_in[0];   // (2, 2048, 2048)
  const float* w_qkv = (const float*)d_in[1];   // (6144, 2048)
  const float* w_op  = (const float*)d_in[2];   // (2048, 2048)
  float* out = (float*)d_out;                   // (2, 2048, 2048)

  const size_t head_elems = (size_t)kB * kNH * kT * kDH;   // 8M bf16
  unsigned short* qh = (unsigned short*)d_ws;
  unsigned short* kh = qh + head_elems;
  unsigned short* vt = kh + head_elems;         // transposed [b][h][d][t]
  unsigned short* ao = vt + head_elems;         // (4096, 2048) bf16

  qkv_gemm_kernel<<<dim3(kN1 / 128, kM / 64), 256, 0, stream>>>(x, w_qkv, qh,
                                                                kh, vt);
  rope_kernel<<<dim3((kB * kNH * kT * 64) / 256, 2), 256, 0, stream>>>(qh, kh);
  flash_attn_kernel<<<dim3(kT / 64, kNH, kB), 128, 0, stream>>>(qh, kh, vt, ao);
  out_proj_kernel<<<dim3(kDM / 128, kM / 64), 256, 0, stream>>>(ao, w_op, out);
}
